// cNet_MNIST_86303072846334
// MI455X (gfx1250) — matrix operand order repaired and checked by reading, not compiled, not run on hardware
//
#include <hip/hip_runtime.h>
#include <hip/hip_bf16.h>

typedef __attribute__((ext_vector_type(16))) _Float16 v16h;
typedef __attribute__((ext_vector_type(8)))  float    v8f;

#define T_STEPS 512
#define MFR_DT  0.1953125f   // 200 * 2^-10
#define THRESH  0.2f
#define DECAY   0.25f

// ---------------- LDS carve (bytes), all sections 32B-aligned where needed --------
#define OFF_PIN    0        // f32[784]   input firing rates
#define OFF_XIN    3136     // f16[784]   input spikes
#define OFF_S1     4704     // f16[2704]  16x13x13
#define OFF_S2     10112    // f16[3872]  32x11x11
#define OFF_S3     17856    // f16[1600]  64x5x5
#define OFF_S4     21056    // f16[40]    10x2x2
#define OFF_DUMP   21136    // f16[32]    dead-store sink
#define OFF_WFC    21216    // f32[400]
#define OFF_KMAP1  22816    // u16[32]
#define OFF_KMAP2  22880    // u16[160]
#define OFF_KMAP3  23200    // u16[288]
#define OFF_KMAP4  23776    // u16[1024]
#define OFF_ASWZ1  25824    // f16[1*1*512]
#define OFF_ASWZ2  26848    // f16[2*5*512]
#define OFF_ASWZ3  37088    // f16[4*9*512]
#define OFF_ASWZ4  73952    // f16[1*32*512]
#define SMEM_BYTES 106752   // end = 106720, rounded up

// ---------- stateless counter-based RNG: u in (0,1) ----------
__device__ __forceinline__ float urand(unsigned a, unsigned b, unsigned c) {
  unsigned h = a * 0x9E3779B1u;
  h ^= b + 0x9E3779B9u + (h << 6) + (h >> 2);
  h ^= c + 0x9E3779B9u + (h << 6) + (h >> 2);
  h ^= h >> 16; h *= 0x7FEB352Du; h ^= h >> 15; h *= 0x846CA68Bu; h ^= h >> 16;
  return (float)(h >> 8) * (1.0f / 16777216.0f);
}

// ---------- one-time: weights -> CDNA5 A-fragment layout (per-lane 32B blocks) ----
// dst[((mTile*nChunks + c)*32 + lane)*16 + e]; zero-padded so no K/M guards at use.
__device__ __forceinline__ void build_aswz(const float* __restrict__ w,
                                           _Float16* __restrict__ dst,
                                           int nTilesM, int nChunks, int Ktot, int Mv,
                                           int tid) {
  const int total = nTilesM * nChunks * 512;
  for (int i = tid; i < total; i += 512) {
    const int e    = i & 15;
    const int lane = (i >> 4) & 31;
    const int tc   = i >> 9;
    const int mTile = tc / nChunks, c = tc - mTile * nChunks;
    const int h = lane >> 4, m = lane & 15;
    const int row = mTile * 16 + m;
    const int k = c * 32 + ((e >> 3) << 4) + (h << 3) + (e & 7);
    _Float16 v = (_Float16)0.0f;
    if (row < Mv && k < Ktot) v = (_Float16)w[row * Ktot + k];
    dst[i] = v;
  }
}

// ---------- one-time: kmap[k] = ic*IHW + ky*IW + kx (time-invariant im2col term) --
template<int KW, int KHW, int IW, int IHW>
__device__ __forceinline__ void build_kmap(unsigned short* __restrict__ km,
                                           int Ktot, int nPad, int tid) {
  for (int k = tid; k < nPad; k += 512) {
    unsigned short v = 0;
    if (k < Ktot) {
      const int ic = k / KHW, r = k - ic * KHW;
      const int ky = r / KW,  kx = r - ky * KW;
      v = (unsigned short)(ic * IHW + ky * IW + kx);
    }
    km[k] = v;
  }
}

// ---------- one 16x16 conv-GEMM tile + two-compartment LIF ----------
template<int OW, int IW, int S>
__device__ __forceinline__ void conv_tile(
    const _Float16* __restrict__ aswz, const unsigned short* __restrict__ kmap,
    const _Float16* __restrict__ in, _Float16* __restrict__ outs,
    _Float16* __restrict__ dump,
    int mBase, int nBase, int nChunks, int Mv, int Nv,
    float* m1, float* m2, float* s,
    unsigned lbase, unsigned seed, unsigned t)
{
  const int lane = (int)(threadIdx.x & 31u);
  const int h = lane >> 4, n = lane & 15;
  const int col = nBase + n;
  const int oy = col / OW, ox = col - oy * OW;
  const _Float16* inb = in + (oy * S) * IW + ox * S;  // per-lane spatial base
  const int mTile = mBase >> 4;
  v8f acc = {};
  for (int c = 0; c < nChunks; ++c) {
    // A: pre-swizzled, contiguous 32B per lane -> 2x ds_load_b128
    v16h a = *(const v16h*)(aswz + (((mTile * nChunks + c) << 5) + lane) * 16);
    // B: gather via time-invariant kmap; no guards (padded A columns are zero,
    // invalid output columns are never stored)
    const unsigned short* km = kmap + (c << 5) + (h << 3);
    v16h b;
#pragma unroll
    for (int e = 0; e < 16; ++e) b[e] = inb[km[(e & 7) + ((e >> 3) << 4)]];
    acc = __builtin_amdgcn_wmma_f32_16x16x32_f16(false, a, false, b,
                                                 (short)0, acc, false, false);
  }
#pragma unroll
  for (int r = 0; r < 8; ++r) {
    const int row = mBase + r + (h << 3);
    const float mm1 = DECAY * m1[r] * (1.0f - s[r]) + acc[r];
    const float mm2 = DECAY * m2[r] + mm1;
    const float u = urand(seed, t, lbase + (unsigned)(row * Nv + col));
    const float ss = ((mm2 - THRESH) * MFR_DT > u) ? 1.0f : 0.0f;
    m1[r] = mm1; m2[r] = mm2; s[r] = ss;
    // branch-free store: invalid elements go to the dump slot
    _Float16* p = (row < Mv && col < Nv) ? (outs + row * Nv + col) : (dump + lane);
    *p = (_Float16)ss;
  }
}

__global__ __launch_bounds__(512, 1) void snn_forward(
    const float* __restrict__ x,  const float* __restrict__ w1,
    const float* __restrict__ w2, const float* __restrict__ w3,
    const float* __restrict__ w4, const float* __restrict__ wfc,
    const int* __restrict__ key_seed, float* __restrict__ out)
{
  extern __shared__ char smem[];
  float*          pIn   = (float*)(smem + OFF_PIN);
  _Float16*       xin   = (_Float16*)(smem + OFF_XIN);
  _Float16*       s1    = (_Float16*)(smem + OFF_S1);
  _Float16*       s2    = (_Float16*)(smem + OFF_S2);
  _Float16*       s3    = (_Float16*)(smem + OFF_S3);
  _Float16*       s4    = (_Float16*)(smem + OFF_S4);
  _Float16*       dump  = (_Float16*)(smem + OFF_DUMP);
  float*          wfcs  = (float*)(smem + OFF_WFC);
  unsigned short* kmap1 = (unsigned short*)(smem + OFF_KMAP1);
  unsigned short* kmap2 = (unsigned short*)(smem + OFF_KMAP2);
  unsigned short* kmap3 = (unsigned short*)(smem + OFF_KMAP3);
  unsigned short* kmap4 = (unsigned short*)(smem + OFF_KMAP4);
  _Float16*       aswz1 = (_Float16*)(smem + OFF_ASWZ1);
  _Float16*       aswz2 = (_Float16*)(smem + OFF_ASWZ2);
  _Float16*       aswz3 = (_Float16*)(smem + OFF_ASWZ3);
  _Float16*       aswz4 = (_Float16*)(smem + OFF_ASWZ4);

  const int tid  = (int)threadIdx.x;
  const int wave = tid >> 5;
  const int lane = tid & 31;
  const int b    = (int)blockIdx.x;
  const unsigned seed =
      (unsigned)key_seed[0] * 0x9E3779B9u ^ ((unsigned)b * 0x85EBCA77u + 0x165667B1u);

  // ---- one-time staging: rates, fc weights, kmaps, swizzled A fragments ----
  const float* xb = x + b * 784;
  for (int i = tid; i < 784; i += 512) pIn[i]  = xb[i] * MFR_DT;
  for (int i = tid; i < 400; i += 512) wfcs[i] = wfc[i];
  build_kmap<4, 16, 28, 784>(kmap1, 16,   32,   tid);
  build_kmap<3, 9,  13, 169>(kmap2, 144,  160,  tid);
  build_kmap<3, 9,  11, 121>(kmap3, 288,  288,  tid);
  build_kmap<4, 16, 5,  25 >(kmap4, 1024, 1024, tid);
  build_aswz(w1, aswz1, 1, 1,  16,   16, tid);
  build_aswz(w2, aswz2, 2, 5,  144,  32, tid);
  build_aswz(w3, aswz3, 4, 9,  288,  64, tid);
  build_aswz(w4, aswz4, 1, 32, 1024, 10, tid);
  __syncthreads();

  // ---- persistent per-lane membrane state (VGPR-resident for all 512 steps) ----
  float c1m1[8] = {}, c1m2[8] = {}, c1s[8] = {};
  float c2m1[8] = {}, c2m2[8] = {}, c2s[8] = {};
  float c3m1[8] = {}, c3m2[8] = {}, c3s[8] = {};
  float c4m1[8] = {}, c4m2[8] = {}, c4s[8] = {};
  float f_m1 = 0.f, f_m2 = 0.f, f_s = 0.f, f_acc = 0.f;

  for (unsigned t = 0; t < T_STEPS; ++t) {
    // rate-coded input spikes
    for (int i = tid; i < 784; i += 512) {
      const float u = urand(seed, t, 0x01000000u + (unsigned)i);
      xin[i] = (_Float16)((pIn[i] > u) ? 1.0f : 0.0f);
    }
    __syncthreads();

    // conv1: M=16 N=169 K=16 (1 chunk). 11 N-tiles -> waves 0..10
    if (wave < 11)
      conv_tile<13, 28, 2>(aswz1, kmap1, xin, s1, dump, 0, wave * 16, 1, 16, 169,
                           c1m1, c1m2, c1s, 0x02000000u, seed, t);
    __syncthreads();

    // conv2: M=32 N=121 K=144 (5 chunks). 2x8 = 16 tiles -> all 16 waves
    conv_tile<11, 13, 1>(aswz2, kmap2, s1, s2, dump, (wave >> 3) * 16, (wave & 7) * 16,
                         5, 32, 121, c2m1, c2m2, c2s, 0x03000000u, seed, t);
    __syncthreads();

    // conv3: M=64 N=25 K=288 (9 chunks). 4x2 = 8 tiles -> waves 0..7
    if (wave < 8)
      conv_tile<5, 11, 2>(aswz3, kmap3, s2, s3, dump, (wave >> 1) * 16, (wave & 1) * 16,
                          9, 64, 25, c3m1, c3m2, c3s, 0x04000000u, seed, t);
    __syncthreads();

    // conv4: M=10(16) N=4(16) K=1024 (32 chunks) -> wave 0
    if (wave == 0)
      conv_tile<2, 5, 1>(aswz4, kmap4, s3, s4, dump, 0, 0, 32, 10, 4,
                         c4m1, c4m2, c4s, 0x05000000u, seed, t);
    __syncthreads();

    // fc (40 -> 10) + output LIF + spike accumulation: wave 0, lanes 0..9
    if (wave == 0 && lane < 10) {
      float d = 0.0f;
#pragma unroll
      for (int j = 0; j < 40; ++j) d += (float)s4[j] * wfcs[lane * 40 + j];
      f_m1 = DECAY * f_m1 * (1.0f - f_s) + d;
      f_m2 = DECAY * f_m2 + f_m1;
      const float u = urand(seed, t, 0x06000000u + (unsigned)lane);
      f_s = ((f_m2 - THRESH) * MFR_DT > u) ? 1.0f : 0.0f;
      f_acc += f_s;
    }
    __syncthreads();
  }

  if (wave == 0 && lane < 10)
    out[b * 10 + lane] = f_acc * (1.0f / (float)T_STEPS);
}

extern "C" void kernel_launch(void* const* d_in, const int* in_sizes, int n_in,
                              void* d_out, int out_size, void* d_ws, size_t ws_size,
                              hipStream_t stream) {
  (void)in_sizes; (void)n_in; (void)out_size; (void)d_ws; (void)ws_size;
  const float* x   = (const float*)d_in[0];
  const float* w1  = (const float*)d_in[1];
  const float* w2  = (const float*)d_in[2];
  const float* w3  = (const float*)d_in[3];
  const float* w4  = (const float*)d_in[4];
  const float* wfc = (const float*)d_in[5];
  const int*   ks  = (const int*)d_in[6];
  snn_forward<<<dim3(32), dim3(512), SMEM_BYTES, stream>>>(
      x, w1, w2, w3, w4, wfc, ks, (float*)d_out);
}
